// GATRecommender_38611755991229
// MI455X (gfx1250) — compile-verified, matrix operand order checked by reading, not run on hardware
//
#include <hip/hip_runtime.h>
#include <hip/hip_bf16.h>
#include <float.h>

// ---------------------------------------------------------------------------
// Problem constants (match reference)
// ---------------------------------------------------------------------------
#define NUM_USERS 100000
#define NUM_ITEMS 50000
#define NUM_NODES (NUM_USERS + NUM_ITEMS)   // 150000 (divisible by 16)
#define EMB       64
#define NUM_EDGES 2000000
#define BATCH     16384

typedef __attribute__((ext_vector_type(16))) __bf16 v16bf;
typedef __attribute__((ext_vector_type(8)))  float  v8f;

// ---------------------------------------------------------------------------
// Helpers
// ---------------------------------------------------------------------------
__device__ __forceinline__ unsigned short f2bf(float x) {
    unsigned u = __float_as_uint(x);
    unsigned r = u + 0x7FFFu + ((u >> 16) & 1u);   // round-to-nearest-even
    return (unsigned short)(r >> 16);
}

__device__ __forceinline__ void atomicMaxFloat(float* addr, float v) {
    if (v >= 0.0f) atomicMax((int*)addr, __float_as_int(v));
    else           atomicMin((unsigned int*)addr, __float_as_uint(v));
}

// ---------------------------------------------------------------------------
// f32 -> bf16 conversion kernels
// ---------------------------------------------------------------------------
__global__ void k_concat_bf16(const float* __restrict__ user,
                              const float* __restrict__ item,
                              unsigned short* __restrict__ out) {
    long long t = (long long)blockIdx.x * blockDim.x + threadIdx.x;
    const long long userN = (long long)NUM_USERS * EMB;
    const long long total = (long long)NUM_NODES * EMB;
    if (t >= total) return;
    float x = (t < userN) ? user[t] : item[t - userN];
    out[t] = f2bf(x);
}

// Convert weight matrix W (K x Ncols, f32 row-major) directly into the
// per-lane WMMA B-fragment layout (bf16):
//   Bp[ ((ct*KS + ks)*32 + lane)*16 + e ] = bf16( W[k][n] )
//   kb = ks*32 + (lane>>4)*8 ; k = kb + (e<8 ? e : e+8) ; n = ct*16 + (lane&15)
__global__ void k_prep_b(const float* __restrict__ W,
                         unsigned short* __restrict__ Bp, int K, int Ncols) {
    int t = blockIdx.x * blockDim.x + threadIdx.x;
    int KS = K >> 5;
    int total = K * Ncols;
    if (t >= total) return;
    int e    = t & 15;
    int lane = (t >> 4) & 31;
    int grp  = t >> 9;              // ct*KS + ks
    int ks   = grp % KS;
    int ct   = grp / KS;
    int kb   = ks * 32 + (lane >> 4) * 8;
    int k    = kb + (e < 8 ? e : e + 8);
    int n    = ct * 16 + (lane & 15);
    Bp[t] = f2bf(W[k * Ncols + n]);
}

// ELU + convert to bf16 (layer-2 input)
__global__ void k_elu_bf16(const float* __restrict__ in,
                           unsigned short* __restrict__ out, long long n) {
    long long t = (long long)blockIdx.x * blockDim.x + threadIdx.x;
    if (t >= n) return;
    float x = in[t];
    float y = (x > 0.0f) ? x : (expf(x) - 1.0f);
    out[t] = f2bf(y);
}

// FINAL = all_emb (residual pre-init; layer-2 messages atomically add on top)
__global__ void k_init_residual(const float* __restrict__ user,
                                const float* __restrict__ item,
                                float* __restrict__ out) {
    long long t = (long long)blockIdx.x * blockDim.x + threadIdx.x;
    const long long userN = (long long)NUM_USERS * EMB;
    const long long total = (long long)NUM_NODES * EMB;
    if (t >= total) return;
    out[t] = (t < userN) ? user[t] : item[t - userN];
}

__global__ void k_init_max(float* g) {          // two global-max scalars
    if (threadIdx.x < 2) g[threadIdx.x] = -FLT_MAX;
}

// ---------------------------------------------------------------------------
// WMMA GEMM:  C[Nrows x COLT*16] = A[Nrows x K] (bf16) * Bp (fragment-ready)
// One wave owns a 16-row strip: loads all K/32 A fragments once into
// registers, then sweeps every 16-wide column tile.  Fully unrolled ->
// COLT*(K/32) back-to-back v_wmma_f32_16x16x32_bf16 per strip.
// ---------------------------------------------------------------------------
template<int K, int COLT>
__global__ void __launch_bounds__(256)
k_gemm_bf16(const unsigned short* __restrict__ A,
            const unsigned short* __restrict__ Bp,
            float* __restrict__ C, int Nrows) {
    constexpr int NC = COLT * 16;
    constexpr int KS = K / 32;
    const int wave = (blockIdx.x * blockDim.x + threadIdx.x) >> 5;
    const int lane = threadIdx.x & 31;
    if (wave * 16 >= Nrows) return;          // wave-uniform exit
    const int m       = lane & 15;
    const int halfSel = lane >> 4;

    union F { v16bf v; uint4 u[2]; };

    // ---- A fragments for this 16-row strip (held in VGPRs, reused COLT x)
    F a[KS];
    const unsigned short* arow = A + ((long long)wave * 16 + m) * K;
    #pragma unroll
    for (int ks = 0; ks < KS; ++ks) {
        const uint4* ap = reinterpret_cast<const uint4*>(arow + ks * 32 + halfSel * 8);
        a[ks].u[0] = ap[0];     // k = kb .. kb+7
        a[ks].u[1] = ap[2];     // k = kb+16 .. kb+23
    }

    // ---- sweep column tiles
    #pragma unroll
    for (int ct = 0; ct < COLT; ++ct) {
        v8f acc = {};
        #pragma unroll
        for (int ks = 0; ks < KS; ++ks) {
            F b;
            const uint4* bp = reinterpret_cast<const uint4*>(
                Bp + (((ct * KS + ks) * 32 + lane) << 4));
            b.u[0] = bp[0];
            b.u[1] = bp[1];
            acc = __builtin_amdgcn_wmma_f32_16x16x32_bf16(
                      false, a[ks].v, false, b.v, (short)0, acc, false, false);
        }
        // store: VGPR r -> row = wave*16 + halfSel*8 + r, col = ct*16 + m
        float* crow = C + ((long long)wave * 16 + halfSel * 8) * NC + ct * 16 + m;
        #pragma unroll
        for (int r = 0; r < 8; ++r) crow[(long long)r * NC] = acc[r];
    }
}

// ---------------------------------------------------------------------------
// Per-(node,head) attention scores: e_src = h . a_l ; e_dst = h . a_r
// H is (N, heads*64); a is (heads, 128)
// ---------------------------------------------------------------------------
__global__ void k_edge_scores(const float* __restrict__ H,
                              const float* __restrict__ a,
                              float* __restrict__ esrc,
                              float* __restrict__ edst,
                              int N, int heads) {
    int t = blockIdx.x * blockDim.x + threadIdx.x;
    if (t >= N * heads) return;
    int n = t / heads, h = t % heads;
    const float* hp = H + (long long)n * (heads * EMB) + h * EMB;
    const float* al = a + h * 2 * EMB;
    const float* ar = al + EMB;
    float s1 = 0.f, s2 = 0.f;
    #pragma unroll 8
    for (int f = 0; f < EMB; ++f) {
        float x = hp[f];
        s1 += x * al[f];
        s2 += x * ar[f];
    }
    esrc[t] = s1;
    edst[t] = s2;
}

// ---------------------------------------------------------------------------
// Edge attention pass 1: leaky-relu logits + global max reduction
// ---------------------------------------------------------------------------
__global__ void __launch_bounds__(256)
k_att_leaky_max(const int* __restrict__ src, const int* __restrict__ dst,
                const float* __restrict__ esrc, const float* __restrict__ edst,
                float* __restrict__ att, float* __restrict__ gmax,
                int heads, long long total) {
    long long t = (long long)blockIdx.x * blockDim.x + threadIdx.x;
    float v = -FLT_MAX;
    if (t < total) {
        int e = (int)(t / heads), h = (int)(t % heads);
        float x = esrc[(long long)src[e] * heads + h] +
                  edst[(long long)dst[e] * heads + h];
        v = (x > 0.0f) ? x : 0.2f * x;
        att[t] = v;
    }
    __shared__ float smax[256];
    smax[threadIdx.x] = v;
    __syncthreads();
    for (int s = 128; s > 0; s >>= 1) {
        if (threadIdx.x < s)
            smax[threadIdx.x] = fmaxf(smax[threadIdx.x], smax[threadIdx.x + s]);
        __syncthreads();
    }
    if (threadIdx.x == 0) atomicMaxFloat(gmax, smax[0]);
}

// Edge attention pass 2: exp(att - max) + segment sum over dst
__global__ void k_att_exp_sum(const int* __restrict__ dst,
                              float* __restrict__ att,
                              float* __restrict__ attsum,
                              const float* __restrict__ gmax,
                              int heads, long long total) {
    long long t = (long long)blockIdx.x * blockDim.x + threadIdx.x;
    if (t >= total) return;
    int e = (int)(t / heads), h = (int)(t % heads);
    float v = expf(att[t] - gmax[0]);
    att[t] = v;
    atomicAdd(&attsum[(long long)dst[e] * heads + h], v);
}

// ---------------------------------------------------------------------------
// Message passing, layer 1: 64 threads/edge, each handles 4 of 256 features
// ---------------------------------------------------------------------------
__global__ void __launch_bounds__(256)
k_message1(const int* __restrict__ src, const int* __restrict__ dst,
           const float* __restrict__ att, const float* __restrict__ attsum,
           const float* __restrict__ H1, float* __restrict__ OUT1,
           long long total) {
    long long t = (long long)blockIdx.x * blockDim.x + threadIdx.x;
    if (t >= total) return;
    int e  = (int)(t >> 6);
    int f4 = (int)(t & 63) * 4;        // 0..252
    int head = f4 >> 6;                // 4 heads of 64 feats
    int s = src[e], d = dst[e];
    const float* hp = H1 + (long long)s * 256 + f4;
    __builtin_prefetch(hp, 0, 3);      // global_prefetch: hide random-row fetch
    float w = att[(long long)e * 4 + head] /
              (attsum[(long long)d * 4 + head] + 1e-8f);
    const float4 hv = *reinterpret_cast<const float4*>(hp);
    float* op = OUT1 + (long long)d * 256 + f4;
    atomicAdd(op + 0, w * hv.x);
    atomicAdd(op + 1, w * hv.y);
    atomicAdd(op + 2, w * hv.z);
    atomicAdd(op + 3, w * hv.w);
}

// Message passing, layer 2: 16 threads/edge, 4 feats each (64 total, 1 head)
__global__ void __launch_bounds__(256)
k_message2(const int* __restrict__ src, const int* __restrict__ dst,
           const float* __restrict__ att, const float* __restrict__ attsum,
           const float* __restrict__ H2, float* __restrict__ FINAL,
           long long total) {
    long long t = (long long)blockIdx.x * blockDim.x + threadIdx.x;
    if (t >= total) return;
    int e  = (int)(t >> 4);
    int f4 = (int)(t & 15) * 4;        // 0..60
    int s = src[e], d = dst[e];
    const float* hp = H2 + (long long)s * 64 + f4;
    __builtin_prefetch(hp, 0, 3);
    float w = att[e] / (attsum[d] + 1e-8f);
    const float4 hv = *reinterpret_cast<const float4*>(hp);
    float* op = FINAL + (long long)d * 64 + f4;
    atomicAdd(op + 0, w * hv.x);
    atomicAdd(op + 1, w * hv.y);
    atomicAdd(op + 2, w * hv.z);
    atomicAdd(op + 3, w * hv.w);
}

// ---------------------------------------------------------------------------
// Final batched dot product: one wave per output element
// ---------------------------------------------------------------------------
__global__ void __launch_bounds__(256)
k_dot(const float* __restrict__ FINAL,
      const int* __restrict__ uid, const int* __restrict__ iid,
      float* __restrict__ out, int batch) {
    int w    = (blockIdx.x * blockDim.x + threadIdx.x) >> 5;
    int lane = threadIdx.x & 31;
    if (w >= batch) return;
    const float* up = FINAL + (long long)uid[w] * 64;
    const float* ip = FINAL + ((long long)NUM_USERS + iid[w]) * 64;
    float s = up[lane] * ip[lane] + up[lane + 32] * ip[lane + 32];
    #pragma unroll
    for (int off = 16; off > 0; off >>= 1) s += __shfl_xor(s, off, 32);
    if (lane == 0) out[w] = s;
}

// ---------------------------------------------------------------------------
// Launcher
// ---------------------------------------------------------------------------
static inline size_t alignUp(size_t x) { return (x + 255) & ~(size_t)255; }

extern "C" void kernel_launch(void* const* d_in, const int* in_sizes, int n_in,
                              void* d_out, int out_size, void* d_ws, size_t ws_size,
                              hipStream_t stream) {
    const float* user_table = (const float*)d_in[0];
    const float* item_table = (const float*)d_in[1];
    const float* W1         = (const float*)d_in[2];
    const float* a1         = (const float*)d_in[3];
    const float* W2         = (const float*)d_in[4];
    const float* a2         = (const float*)d_in[5];
    const int*   edge_index = (const int*)d_in[6];
    const int*   user_ids   = (const int*)d_in[7];
    const int*   item_ids   = (const int*)d_in[8];
    float*       out        = (float*)d_out;

    const int*  src = edge_index;               // edge_index[0], E ints
    const int*  dst = edge_index + NUM_EDGES;   // edge_index[1]

    const long long N = NUM_NODES, E = NUM_EDGES;

    // --- workspace sub-allocation ------------------------------------------
    char* ws = (char*)d_ws;
    size_t off = 0;
    #define WSALLOC(ptr, type, count) \
        type* ptr = (type*)(ws + off); off = alignUp(off + (size_t)(count) * sizeof(type));
    WSALLOC(A_bf,    unsigned short, N * EMB)        // bf16 all_emb
    WSALLOC(W1p,     unsigned short, EMB * 256)      // fragment-ready W1
    WSALLOC(W2p,     unsigned short, 256 * EMB)      // fragment-ready W2
    WSALLOC(H1,      float,          N * 256)        // layer-1 features (reused for H2)
    WSALLOC(OUT1,    float,          N * 256)        // layer-1 aggregate (reused for FINAL)
    WSALLOC(X2_bf,   unsigned short, N * 256)        // elu(OUT1) in bf16
    WSALLOC(esrc1,   float,          N * 4)
    WSALLOC(edst1,   float,          N * 4)
    WSALLOC(att1,    float,          E * 4)
    WSALLOC(attsum1, float,          N * 4)
    WSALLOC(esrc2,   float,          N)
    WSALLOC(edst2,   float,          N)
    WSALLOC(att2,    float,          E)
    WSALLOC(attsum2, float,          N)
    WSALLOC(gmax,    float,          64)
    #undef WSALLOC
    (void)ws_size;
    float* H2    = H1;     // reuse: H1 dead after layer-1 message pass
    float* FINAL = OUT1;   // reuse: OUT1 dead after X2 conversion

    const int TB = 256;
    #define NB(total) ((int)(((long long)(total) + TB - 1) / TB))

    // --- conversions & init -------------------------------------------------
    k_concat_bf16 <<<NB(N * EMB), TB, 0, stream>>>(user_table, item_table, A_bf);
    k_prep_b      <<<NB(EMB * 256), TB, 0, stream>>>(W1, W1p, EMB, 256);
    k_prep_b      <<<NB(256 * EMB), TB, 0, stream>>>(W2, W2p, 256, EMB);
    k_init_max    <<<1, 32, 0, stream>>>(gmax);
    hipMemsetAsync(OUT1,    0, (size_t)N * 256 * sizeof(float), stream);
    hipMemsetAsync(attsum1, 0, (size_t)N * 4   * sizeof(float), stream);
    hipMemsetAsync(attsum2, 0, (size_t)N       * sizeof(float), stream);

    // --- layer 1: GEMM (WMMA), scores, attention, messages ------------------
    {
        long long waves = N / 16;                    // one wave per 16-row strip
        k_gemm_bf16<64, 16> <<<NB(waves * 32), TB, 0, stream>>>(A_bf, W1p, H1, (int)N);
    }
    k_edge_scores <<<NB(N * 4), TB, 0, stream>>>(H1, a1, esrc1, edst1, (int)N, 4);
    k_att_leaky_max <<<NB(E * 4), TB, 0, stream>>>(src, dst, esrc1, edst1,
                                                   att1, gmax, 4, E * 4);
    k_att_exp_sum <<<NB(E * 4), TB, 0, stream>>>(dst, att1, attsum1, gmax, 4, E * 4);
    k_message1    <<<NB(E * 64), TB, 0, stream>>>(src, dst, att1, attsum1,
                                                  H1, OUT1, E * 64);

    // --- ELU -> bf16 (layer-2 input) ----------------------------------------
    k_elu_bf16 <<<NB(N * 256), TB, 0, stream>>>(OUT1, X2_bf, N * 256);

    // --- residual init must follow OUT1 consumption (FINAL aliases OUT1) ----
    k_init_residual <<<NB(N * EMB), TB, 0, stream>>>(user_table, item_table, FINAL);

    // --- layer 2: GEMM (WMMA), scores, attention, messages ------------------
    {
        long long waves = N / 16;
        k_gemm_bf16<256, 4> <<<NB(waves * 32), TB, 0, stream>>>(X2_bf, W2p, H2, (int)N);
    }
    k_edge_scores <<<NB(N), TB, 0, stream>>>(H2, a2, esrc2, edst2, (int)N, 1);
    k_att_leaky_max <<<NB(E), TB, 0, stream>>>(src, dst, esrc2, edst2,
                                               att2, gmax + 1, 1, E);
    k_att_exp_sum <<<NB(E), TB, 0, stream>>>(dst, att2, attsum2, gmax + 1, 1, E);
    k_message2    <<<NB(E * 16), TB, 0, stream>>>(src, dst, att2, attsum2,
                                                  H2, FINAL, E * 16);

    // --- final batched dot ---------------------------------------------------
    k_dot <<<NB((long long)BATCH * 32), TB, 0, stream>>>(FINAL, user_ids, item_ids,
                                                         out, BATCH);
    #undef NB
}
